// GraphNetwork_38113539784807
// MI455X (gfx1250) — matrix operand order repaired and checked by reading, not compiled, not run on hardware
//
#include <hip/hip_runtime.h>
#include <hip/hip_bf16.h>
#include <math.h>

#define NN 8192   // nodes
#define NE 32768  // edges
#define NB 256    // batches
#define BIGC 0x40000000

typedef __attribute__((ext_vector_type(16))) __bf16 v16bf;
typedef __attribute__((ext_vector_type(8)))  float  v8f;
typedef __attribute__((ext_vector_type(4)))  unsigned int v4u;
typedef __attribute__((ext_vector_type(4)))  unsigned short v4us;
typedef __attribute__((ext_vector_type(4)))  float v4f;

union Frag16 { v16bf v; v4u q[2]; };

__device__ __forceinline__ unsigned short f2bf(float f) {
  unsigned int u = __float_as_uint(f);
  u += 0x7FFFu + ((u >> 16) & 1u);   // round-to-nearest-even
  return (unsigned short)(u >> 16);
}
__device__ __forceinline__ float gelu_exact(float x) {
  return 0.5f * x * (1.0f + erff(x * 0.70710678118654752f));
}
__device__ __forceinline__ void atomicMaxF(float* a, float v) {
  if (v >= 0.0f) atomicMax((int*)a, __float_as_int(v));
  else           atomicMin((unsigned int*)a, __float_as_uint(v));
}

// ---------------- token pool: mean over tokens[:,1:16] of emb rows, masked PAD/SEP
__global__ __launch_bounds__(256) void token_pool_kernel(
    const int* __restrict__ toks, const float* __restrict__ emb, float* __restrict__ out) {
  const int row = blockIdx.x;
  const int c = threadIdx.x;          // 256 threads cover 768 cols, 3 each
  float a0 = 0.f, a1 = 0.f, a2 = 0.f;
  #pragma unroll
  for (int t = 1; t < 16; ++t) {
    int tok = toks[row * 16 + t];
    float m = (tok == 1 || tok == 2) ? 0.0f : 1.0f;   // PAD=1, SEP=2
    const float* e = emb + (long)tok * 768;
    a0 += m * e[c]; a1 += m * e[c + 256]; a2 += m * e[c + 512];
  }
  const float s = 1.0f / 15.0f;
  float* o = out + (long)row * 768;
  o[c] = a0 * s; o[c + 256] = a1 * s; o[c + 512] = a2 * s;
}

// ---------------- weight prep: Wt[n*K + k] = bf16(W[k*Nout + n])  (transposed, bf16)
__global__ void wt_bf16_kernel(const float* __restrict__ W, unsigned short* __restrict__ Wt,
                               int K, int Nout) {
  int i = blockIdx.x * blockDim.x + threadIdx.x;
  if (i < K * Nout) {
    int k = i / Nout, n = i - k * Nout;      // coalesced read over n
    Wt[(long)n * K + k] = f2bf(W[i]);
  }
}

// ---------------- WMMA bf16 GEMM, fused segment-gather A, async-LDS W, optional LN+GELU
// A row e = concat of up to 4 f32 segments. Wt is bf16, pre-transposed [Nout][K].
// Block: 128 threads (4 waves). Tile: 64 rows x 128 cols. K step 32, ping-pong LDS.
__global__ __launch_bounds__(128) void gemm_bf16_kernel(
    const float* __restrict__ src0, const int* __restrict__ idx0, int w0,
    const float* __restrict__ src1, const int* __restrict__ idx1, int w1,
    const float* __restrict__ src2, const int* __restrict__ idx2, int w2,
    const float* __restrict__ src3, const int* __restrict__ idx3, int w3,
    int c1, int c2, int c3, int K,
    const unsigned short* __restrict__ Wt, const float* __restrict__ bias, int Nout,
    const float* __restrict__ gamma, const float* __restrict__ beta, int doLN,
    float* __restrict__ out) {
  __shared__ unsigned short ldsA[2][64 * 32];    // 2 x 4 KB  A tiles (bf16)
  __shared__ unsigned short ldsW[2][128 * 32];   // 2 x 8 KB  W tiles [n][k] (bf16)
  __shared__ float ldsOut[64 * 128];             // 32 KB     LN staging

  const int tid = threadIdx.x;
  const int r0 = blockIdx.x * 64;
  const int n0 = blockIdx.y * 128;

  const int lane = tid & 31;
  const int wv = tid >> 5;
  const int lhalf = lane >> 4;   // 0 or 1
  const int l15 = lane & 15;
  const int rl = tid >> 1;       // staging row 0..63
  const int kc = (tid & 1) * 16; // staging k offset

  // per-thread gather offsets for this thread's staging row (registers, no LDS table)
  const int e = r0 + rl;
  const int ro0 = (idx0 ? idx0[e] : e) * w0;
  const int ro1 = (idx1 ? idx1[e] : e) * w1;
  const int ro2 = (idx2 ? idx2[e] : e) * w2;
  const int ro3 = (idx3 ? idx3[e] : e) * w3;

  v8f acc[8];
  #pragma unroll
  for (int c = 0; c < 8; ++c) { v8f z = {0,0,0,0,0,0,0,0}; acc[c] = z; }

  auto stage = [&](int kk, int buf) {
    // A tile: segment is uniform in kk (all boundaries are multiples of 32) -> scalar branch
    const float* sp; int ro;
    if (kk < c1)      { sp = src0; ro = ro0 + kk; }
    else if (kk < c2) { sp = src1; ro = ro1 + (kk - c1); }
    else if (kk < c3) { sp = src2; ro = ro2 + (kk - c2); }
    else              { sp = src3; ro = ro3 + (kk - c3); }
    const v4f* p4 = (const v4f*)(sp + ro + kc);
    unsigned short* dst = &ldsA[buf][rl * 32 + kc];
    #pragma unroll
    for (int q = 0; q < 4; ++q) {
      v4f f = p4[q];
      v4us h; h.x = f2bf(f.x); h.y = f2bf(f.y); h.z = f2bf(f.z); h.w = f2bf(f.w);
      *(v4us*)(dst + q * 4) = h;
    }
    // W tile: async global->LDS copy, 64 bytes per thread (row n = n0+tid of Wt)
    unsigned lds  = (unsigned)(size_t)&ldsW[buf][tid * 32];
    unsigned gofs = (unsigned)(((long)(n0 + tid) * K + kk) * 2);
    asm volatile(
        "global_load_async_to_lds_b128 %0, %1, %2\n\t"
        "global_load_async_to_lds_b128 %0, %1, %2 offset:16\n\t"
        "global_load_async_to_lds_b128 %0, %1, %2 offset:32\n\t"
        "global_load_async_to_lds_b128 %0, %1, %2 offset:48"
        :: "v"(lds), "v"(gofs), "s"(Wt) : "memory");
  };

  stage(0, 0);
  const int nsteps = K >> 5;
  for (int s = 0; s < nsteps; ++s) {
    asm volatile("s_wait_asynccnt 0" ::: "memory");  // our W tile landed in LDS
    __syncthreads();                                 // everyone's tiles visible
    if (s + 1 < nsteps) stage((s + 1) << 5, (s + 1) & 1);
    const int cur = s & 1;

    // A fragment (ISA 16-bit A 16x32): lane M=l15, K = 8*lhalf..+7 and 16+8*lhalf..+7
    Frag16 af;
    const v4u* pa = (const v4u*)(&ldsA[cur][(wv * 16 + l15) * 32]);
    af.q[0] = pa[lhalf];
    af.q[1] = pa[2 + lhalf];

    // preload ALL 8 B fragments into distinct registers, then back-to-back WMMAs
    Frag16 bf8[8];
    #pragma unroll
    for (int c = 0; c < 8; ++c) {
      const v4u* pb = (const v4u*)(&ldsW[cur][(c * 16 + l15) * 32]);
      bf8[c].q[0] = pb[lhalf];
      bf8[c].q[1] = pb[2 + lhalf];
    }
    #pragma unroll
    for (int c = 0; c < 8; ++c)
      acc[c] = __builtin_amdgcn_wmma_f32_16x16x32_bf16(
          false, af.v, false, bf8[c].v, (short)0, acc[c], false, false);
  }

  if (doLN) {  // Nout==128, gridDim.y==1: bias -> LayerNorm -> exact GELU
    #pragma unroll
    for (int c = 0; c < 8; ++c)
      #pragma unroll
      for (int r = 0; r < 8; ++r)
        ldsOut[(wv * 16 + r + 8 * lhalf) * 128 + c * 16 + l15] = acc[c][r];
    __syncthreads();
    if (tid < 64) {
      float* rowp = ldsOut + tid * 128;
      float mu = 0.f;
      for (int n = 0; n < 128; ++n) { float v = rowp[n] + bias[n]; rowp[n] = v; mu += v; }
      mu *= (1.0f / 128.0f);
      float var = 0.f;
      for (int n = 0; n < 128; ++n) { float d = rowp[n] - mu; var += d * d; }
      var *= (1.0f / 128.0f);
      float rs = rsqrtf(var + 1e-5f);
      float* op = out + (long)(r0 + tid) * Nout;
      for (int n = 0; n < 128; ++n)
        op[n] = gelu_exact((rowp[n] - mu) * rs * gamma[n] + beta[n]);
    }
  } else {     // plain bias + store
    #pragma unroll
    for (int c = 0; c < 8; ++c) {
      int n = n0 + c * 16 + l15;
      float b = bias[n];
      #pragma unroll
      for (int r = 0; r < 8; ++r) {
        int row = r0 + wv * 16 + r + 8 * lhalf;
        out[(long)row * Nout + n] = acc[c][r] + b;
      }
    }
  }
}

// ---------------- weight MLP final layer: (E,128) @ (128,1) + b
__global__ __launch_bounds__(256) void dot128_kernel(
    const float* __restrict__ h, const float* __restrict__ W,
    const float* __restrict__ b, float* __restrict__ outw) {
  int lane = threadIdx.x & 31;
  int e = blockIdx.x * 8 + (threadIdx.x >> 5);
  const float* hp = h + (long)e * 128;
  float s = 0.f;
  #pragma unroll
  for (int k = 0; k < 4; ++k) s += hp[lane + k * 32] * W[lane + k * 32];
  #pragma unroll
  for (int off = 16; off > 0; off >>= 1) s += __shfl_xor(s, off, 32);
  if (lane == 0) outw[e] = s + b[0];
}

// ---------------- misc small kernels
__global__ void fill_kernel(float* p, float v, int n) {
  int i = blockIdx.x * blockDim.x + threadIdx.x;
  if (i < n) p[i] = v;
}
__global__ void edge_batch_kernel(const int* __restrict__ row, const int* __restrict__ nb,
                                  int* __restrict__ eb, int n) {
  int e = blockIdx.x * blockDim.x + threadIdx.x;
  if (e < n) eb[e] = nb[row[e]];
}
__global__ void seg_max_kernel(const float* __restrict__ w, const int* __restrict__ eb,
                               float* __restrict__ bmax, int n) {
  int e = blockIdx.x * blockDim.x + threadIdx.x;
  if (e < n) atomicMaxF(&bmax[eb[e]], w[e]);
}
__global__ void seg_expsum_kernel(const float* __restrict__ w, const int* __restrict__ eb,
                                  const float* __restrict__ bmax, float* __restrict__ ex,
                                  float* __restrict__ bsum, int n) {
  int e = blockIdx.x * blockDim.x + threadIdx.x;
  if (e < n) { float t = expf(w[e] - bmax[eb[e]]); ex[e] = t; atomicAdd(&bsum[eb[e]], t); }
}
__global__ void seg_norm_kernel(float* __restrict__ nw, const int* __restrict__ eb,
                                const float* __restrict__ bsum, int n) {
  int e = blockIdx.x * blockDim.x + threadIdx.x;
  if (e < n) nw[e] = nw[e] / bsum[eb[e]];
}
__global__ __launch_bounds__(256) void recv_scatter_kernel(
    const float* __restrict__ msg, const float* __restrict__ nw,
    const int* __restrict__ col, float* __restrict__ recv, float* __restrict__ cnt) {
  int e = blockIdx.x, c = threadIdx.x;
  int d = col[e]; float w = nw[e];
  atomicAdd(&recv[(long)d * 256 + c], msg[(long)e * 256 + c] * w);
  if (c == 0) atomicAdd(&cnt[d], 1.0f);
}
__global__ void recv_div_kernel(float* __restrict__ recv, const float* __restrict__ cnt, int n) {
  int i = blockIdx.x * blockDim.x + threadIdx.x;
  if (i < n) recv[i] /= fmaxf(cnt[i >> 8], 1.0f);
}
__global__ __launch_bounds__(128) void pool_edge_kernel(
    const float* __restrict__ eo, const float* __restrict__ nw, const int* __restrict__ eb,
    float* __restrict__ pe, float* __restrict__ ceb) {
  int e = blockIdx.x; int b = eb[e]; float w = nw[e];
  for (int c = threadIdx.x; c < 384; c += 128)
    atomicAdd(&pe[(long)b * 384 + c], eo[(long)e * 384 + c] * w);
  if (threadIdx.x == 0) atomicAdd(&ceb[b], 1.0f);
}
__global__ __launch_bounds__(128) void pool_node_kernel(
    const float* __restrict__ xn, const int* __restrict__ nbat,
    float* __restrict__ pn, float* __restrict__ cnb) {
  int i = blockIdx.x; int b = nbat[i];
  for (int c = threadIdx.x; c < 384; c += 128)
    atomicAdd(&pn[(long)b * 384 + c], xn[(long)i * 384 + c]);
  if (threadIdx.x == 0) atomicAdd(&cnb[b], 1.0f);
}
__global__ void assemble_kernel(const float* __restrict__ pe, const float* __restrict__ ceb,
                                const float* __restrict__ pn, const float* __restrict__ cnb,
                                const float* __restrict__ u, const float* __restrict__ nw,
                                float* __restrict__ out) {
  int i = blockIdx.x * blockDim.x + threadIdx.x;
  const int GV = NB * 768;
  if (i < GV) {
    int b = i / 768, c = i % 768;
    out[i] = (c < 384) ? pe[b * 384 + c] / fmaxf(ceb[b], 1.0f)
                       : pn[b * 384 + (c - 384)] / fmaxf(cnb[b], 1.0f);
  } else if (i < 2 * GV) {
    out[i] = u[i - GV];
  } else if (i < 2 * GV + NE) {
    out[i] = nw[i - 2 * GV];
  }
}

// ---------------- host orchestration
extern "C" void kernel_launch(void* const* d_in, const int* in_sizes, int n_in,
                              void* d_out, int out_size, void* d_ws, size_t ws_size,
                              hipStream_t stream) {
  const int* x_tokens    = (const int*)d_in[0];
  const int* edge_tokens = (const int*)d_in[1];
  const int* erow        = (const int*)d_in[2];
  const int* ecol        = erow + NE;
  const float* u         = (const float*)d_in[3];
  const int* node_batch  = (const int*)d_in[4];
  const float* emb       = (const float*)d_in[5];
  const float *eW0=(const float*)d_in[6],  *eb0=(const float*)d_in[7],  *eg0=(const float*)d_in[8],  *ebe0=(const float*)d_in[9];
  const float *eW1=(const float*)d_in[10], *eb1=(const float*)d_in[11], *eg1=(const float*)d_in[12], *ebe1=(const float*)d_in[13];
  const float *eW2=(const float*)d_in[14], *eb2=(const float*)d_in[15];
  const float *wW0=(const float*)d_in[16], *wb0=(const float*)d_in[17], *wg0=(const float*)d_in[18], *wbe0=(const float*)d_in[19];
  const float *wW1=(const float*)d_in[20], *wb1=(const float*)d_in[21];
  const float *mW0=(const float*)d_in[22], *mb0=(const float*)d_in[23], *mg0=(const float*)d_in[24], *mbe0=(const float*)d_in[25];
  const float *mW1=(const float*)d_in[26], *mb1=(const float*)d_in[27], *mg1=(const float*)d_in[28], *mbe1=(const float*)d_in[29];
  const float *mW2=(const float*)d_in[30], *mb2=(const float*)d_in[31];
  const float *nW0=(const float*)d_in[32], *nb0=(const float*)d_in[33], *ng0=(const float*)d_in[34], *nbe0=(const float*)d_in[35];
  const float *nW1=(const float*)d_in[36], *nb1=(const float*)d_in[37], *ng1=(const float*)d_in[38], *nbe1=(const float*)d_in[39];
  const float *nW2=(const float*)d_in[40], *nb2=(const float*)d_in[41];
  float* out = (float*)d_out;

  char* p = (char*)d_ws;
  auto carve = [&](size_t bytes) -> void* {
    void* r = (void*)p; p += (bytes + 255) & ~(size_t)255; return r;
  };
  float* x    = (float*)carve((size_t)NN * 768 * 4);
  float* ea   = (float*)carve((size_t)NE * 768 * 4);
  float* hA   = (float*)carve((size_t)NE * 128 * 4);
  float* hB   = (float*)carve((size_t)NE * 128 * 4);
  float* eout = (float*)carve((size_t)NE * 384 * 4);
  float* msg  = (float*)carve((size_t)NE * 256 * 4);
  float* wts  = (float*)carve((size_t)NE * 4);
  float* nw   = (float*)carve((size_t)NE * 4);
  float* recv = (float*)carve((size_t)NN * 256 * 4);
  float* xnew = (float*)carve((size_t)NN * 384 * 4);
  int*   eb   = (int*)  carve((size_t)NE * 4);
  float* cntc = (float*)carve((size_t)NN * 4);
  float* bmax = (float*)carve((size_t)NB * 4);
  float* bsum = (float*)carve((size_t)NB * 4);
  float* pe   = (float*)carve((size_t)NB * 384 * 4);
  float* ceb  = (float*)carve((size_t)NB * 4);
  float* pn   = (float*)carve((size_t)NB * 384 * 4);
  float* cnb  = (float*)carve((size_t)NB * 4);
  // bf16 transposed weights
  unsigned short* wW0t = (unsigned short*)carve((size_t)128 * 1536 * 2);
  unsigned short* eW0t = (unsigned short*)carve((size_t)128 * 3072 * 2);
  unsigned short* eW1t = (unsigned short*)carve((size_t)128 * 128 * 2);
  unsigned short* eW2t = (unsigned short*)carve((size_t)384 * 128 * 2);
  unsigned short* mW0t = (unsigned short*)carve((size_t)128 * 1152 * 2);
  unsigned short* mW1t = (unsigned short*)carve((size_t)128 * 128 * 2);
  unsigned short* mW2t = (unsigned short*)carve((size_t)256 * 128 * 2);
  unsigned short* nW0t = (unsigned short*)carve((size_t)128 * 1792 * 2);
  unsigned short* nW1t = (unsigned short*)carve((size_t)128 * 128 * 2);
  unsigned short* nW2t = (unsigned short*)carve((size_t)384 * 128 * 2);

  // init accumulators (ws is poisoned between runs)
  fill_kernel<<<(NB + 255) / 256, 256, 0, stream>>>(bmax, -INFINITY, NB);
  fill_kernel<<<(NB + 255) / 256, 256, 0, stream>>>(bsum, 0.f, NB);
  fill_kernel<<<(NN * 256 + 255) / 256, 256, 0, stream>>>(recv, 0.f, NN * 256);
  fill_kernel<<<(NN + 255) / 256, 256, 0, stream>>>(cntc, 0.f, NN);
  fill_kernel<<<(NB * 384 + 255) / 256, 256, 0, stream>>>(pe, 0.f, NB * 384);
  fill_kernel<<<(NB + 255) / 256, 256, 0, stream>>>(ceb, 0.f, NB);
  fill_kernel<<<(NB * 384 + 255) / 256, 256, 0, stream>>>(pn, 0.f, NB * 384);
  fill_kernel<<<(NB + 255) / 256, 256, 0, stream>>>(cnb, 0.f, NB);

  // weight prep (bf16 transposed)
  auto WT = [&](const float* W, unsigned short* Wt, int K, int Nn) {
    wt_bf16_kernel<<<(K * Nn + 255) / 256, 256, 0, stream>>>(W, Wt, K, Nn);
  };
  WT(wW0, wW0t, 1536, 128); WT(eW0, eW0t, 3072, 128); WT(eW1, eW1t, 128, 128);
  WT(eW2, eW2t, 128, 384);  WT(mW0, mW0t, 1152, 128); WT(mW1, mW1t, 128, 128);
  WT(mW2, mW2t, 128, 256);  WT(nW0, nW0t, 1792, 128); WT(nW1, nW1t, 128, 128);
  WT(nW2, nW2t, 128, 384);

  // token pools + edge batch
  token_pool_kernel<<<NN, 256, 0, stream>>>(x_tokens, emb, x);
  token_pool_kernel<<<NE, 256, 0, stream>>>(edge_tokens, emb, ea);
  edge_batch_kernel<<<(NE + 255) / 256, 256, 0, stream>>>(erow, node_batch, eb, NE);

  // weight MLP: [ea, u_e] (1536) -> 128 LN/GELU -> 1
  gemm_bf16_kernel<<<dim3(NE / 64, 1), 128, 0, stream>>>(
      ea, nullptr, 768,  u, eb, 768,  nullptr, nullptr, 0,  nullptr, nullptr, 0,
      768, BIGC, BIGC, 1536, wW0t, wb0, 128, wg0, wbe0, 1, hA);
  dot128_kernel<<<NE / 8, 256, 0, stream>>>(hA, wW1, wb1, wts);

  // scatter softmax over edge_batch
  seg_max_kernel<<<(NE + 255) / 256, 256, 0, stream>>>(wts, eb, bmax, NE);
  seg_expsum_kernel<<<(NE + 255) / 256, 256, 0, stream>>>(wts, eb, bmax, nw, bsum, NE);
  seg_norm_kernel<<<(NE + 255) / 256, 256, 0, stream>>>(nw, eb, bsum, NE);

  // edge MLP: [x[row], x[col], ea, u_e] (3072) -> 128 -> 128 -> 384
  gemm_bf16_kernel<<<dim3(NE / 64, 1), 128, 0, stream>>>(
      x, erow, 768,  x, ecol, 768,  ea, nullptr, 768,  u, eb, 768,
      768, 1536, 2304, 3072, eW0t, eb0, 128, eg0, ebe0, 1, hA);
  gemm_bf16_kernel<<<dim3(NE / 64, 1), 128, 0, stream>>>(
      hA, nullptr, 128, nullptr, nullptr, 0, nullptr, nullptr, 0, nullptr, nullptr, 0,
      BIGC, BIGC, BIGC, 128, eW1t, eb1, 128, eg1, ebe1, 1, hB);
  gemm_bf16_kernel<<<dim3(NE / 64, 3), 128, 0, stream>>>(
      hB, nullptr, 128, nullptr, nullptr, 0, nullptr, nullptr, 0, nullptr, nullptr, 0,
      BIGC, BIGC, BIGC, 128, eW2t, eb2, 384, nullptr, nullptr, 0, eout);

  // msg MLP: [x[row], edge_out] (1152) -> 128 -> 128 -> 256
  gemm_bf16_kernel<<<dim3(NE / 64, 1), 128, 0, stream>>>(
      x, erow, 768,  eout, nullptr, 384,  nullptr, nullptr, 0,  nullptr, nullptr, 0,
      768, BIGC, BIGC, 1152, mW0t, mb0, 128, mg0, mbe0, 1, hA);
  gemm_bf16_kernel<<<dim3(NE / 64, 1), 128, 0, stream>>>(
      hA, nullptr, 128, nullptr, nullptr, 0, nullptr, nullptr, 0, nullptr, nullptr, 0,
      BIGC, BIGC, BIGC, 128, mW1t, mb1, 128, mg1, mbe1, 1, hB);
  gemm_bf16_kernel<<<dim3(NE / 64, 2), 128, 0, stream>>>(
      hB, nullptr, 128, nullptr, nullptr, 0, nullptr, nullptr, 0, nullptr, nullptr, 0,
      BIGC, BIGC, BIGC, 128, mW2t, mb2, 256, nullptr, nullptr, 0, msg);

  // recv = scatter_mean(msg * nw, col, N)
  recv_scatter_kernel<<<NE, 256, 0, stream>>>(msg, nw, ecol, recv, cntc);
  recv_div_kernel<<<(NN * 256 + 255) / 256, 256, 0, stream>>>(recv, cntc, NN * 256);

  // node MLP: [x, recv, u[node_batch]] (1792) -> 128 -> 128 -> 384
  gemm_bf16_kernel<<<dim3(NN / 64, 1), 128, 0, stream>>>(
      x, nullptr, 768,  recv, nullptr, 256,  u, node_batch, 768,  nullptr, nullptr, 0,
      768, 1024, BIGC, 1792, nW0t, nb0, 128, ng0, nbe0, 1, hA);
  gemm_bf16_kernel<<<dim3(NN / 64, 1), 128, 0, stream>>>(
      hA, nullptr, 128, nullptr, nullptr, 0, nullptr, nullptr, 0, nullptr, nullptr, 0,
      BIGC, BIGC, BIGC, 128, nW1t, nb1, 128, ng1, nbe1, 1, hB);
  gemm_bf16_kernel<<<dim3(NN / 64, 3), 128, 0, stream>>>(
      hB, nullptr, 128, nullptr, nullptr, 0, nullptr, nullptr, 0, nullptr, nullptr, 0,
      BIGC, BIGC, BIGC, 128, nW2t, nb2, 384, nullptr, nullptr, 0, xnew);

  // graph pools + output assembly
  pool_edge_kernel<<<NE, 128, 0, stream>>>(eout, nw, eb, pe, ceb);
  pool_node_kernel<<<NN, 128, 0, stream>>>(xnew, node_batch, pn, cnb);
  int total = 2 * NB * 768 + NE;
  assemble_kernel<<<(total + 255) / 256, 256, 0, stream>>>(pe, ceb, pn, cnb, u, nw, out);
}
